// GraphormerFishAttention_45844480918018
// MI455X (gfx1250) — compile-verified, matrix operand order checked by reading, not run on hardware
//
#include <hip/hip_runtime.h>

// ---------------------------------------------------------------------------
// Types for gfx1250 WMMA (wave32). Plain ext_vector types (trivially
// constructible -> legal union members, unlike HIP's uint4 struct).
// ---------------------------------------------------------------------------
typedef __attribute__((ext_vector_type(16))) __bf16    v16bf;
typedef __attribute__((ext_vector_type(8)))  float     v8f;
typedef __attribute__((ext_vector_type(4)))  unsigned  v4u;

union Frag16 { v4u q[2]; v16bf v; };

// Round-to-nearest-even f32 -> bf16
__device__ __forceinline__ __bf16 f2bf(float f) {
    union { float f; unsigned u; } in; in.f = f;
    unsigned u = in.u;
    unsigned r = u + 0x7FFFu + ((u >> 16) & 1u);
    if ((u & 0x7F800000u) == 0x7F800000u) r = u;   // inf/nan: truncate
    union { unsigned short s; __bf16 b; } out; out.s = (unsigned short)(r >> 16);
    return out.b;
}

// Load a 16-element bf16 WMMA fragment (A or B side) from a row-major matrix.
// Per ISA 16-bit layout: lane L holds row (L&15); K elements are
//   e[0..7]  = [kb + 0..7],  e[8..15] = [kb + 16..23],  kb = (L>=16) ? 8 : 0.
// `p` must already point at  base + row*ld + kk + kb  (16B aligned).
__device__ __forceinline__ v16bf load_frag(const __bf16* p) {
    Frag16 f;
    f.q[0] = *reinterpret_cast<const v4u*>(p);
    f.q[1] = *reinterpret_cast<const v4u*>(p + 16);
    return f.v;
}

// ---------------------------------------------------------------------------
// Kernel 1a/1b: f32 -> bf16 convert (plain, and transposed for weights so the
// WMMA B-fragment reads contiguous K).
// ---------------------------------------------------------------------------
__global__ void cvt_bf16(const float* __restrict__ in, __bf16* __restrict__ out, int n) {
    int i = blockIdx.x * blockDim.x + threadIdx.x;
    if (i < n) out[i] = f2bf(in[i]);
}

__global__ void cvt_t_bf16(const float* __restrict__ in, __bf16* __restrict__ out,
                           int rows, int cols) {
    int i = blockIdx.x * blockDim.x + threadIdx.x;
    if (i < rows * cols) {
        int r = i / cols, c = i - r * cols;
        out[c * rows + r] = f2bf(in[i]);   // out is [cols][rows] = B^T
    }
}

// ---------------------------------------------------------------------------
// Kernel 2: generic bf16 WMMA GEMM  C[M,N] = A[M,K] @ B[K,N] (+bias)
//   A row-major (lda), BT = B^T row-major [N][K] (ldb).
//   Block = 256 threads = 8 waves in 2x4 -> 32x128 block tile.
//   Each wave: one A fragment x two B fragments -> two independent WMMA
//   chains (latency hiding + A-load reuse).
//   mode 0: bf16 store C[m*ldc+n]           (Q,K projections)
//   mode 1: +bias, write V transposed per (b,l) head: Vt[b][l][d][t]
//   mode 2: +bias, f32 store to out[m*ldc+n] (final Wout GEMM)
// ---------------------------------------------------------------------------
__global__ void gemm_bf16_wmma(const __bf16* __restrict__ A, int lda,
                               const __bf16* __restrict__ BT, int ldb,
                               int K, const float* __restrict__ bias,
                               float* __restrict__ outF, __bf16* __restrict__ outBF,
                               int mode, int ldc) {
    const int lane = threadIdx.x & 31;
    const int wid  = threadIdx.x >> 5;
    const int m0   = blockIdx.x * 32  + (wid >> 2) * 16;
    const int n0   = blockIdx.y * 128 + (wid & 3) * 32;   // wave covers n0..n0+31
    const int kb   = (lane & 16) ? 8 : 0;
    const __bf16* Ap  = A  + (size_t)(m0 + (lane & 15)) * lda + kb;
    const __bf16* Bp0 = BT + (size_t)(n0 + (lane & 15)) * ldb + kb;
    const __bf16* Bp1 = Bp0 + (size_t)16 * ldb;

    v8f acc0 = {}, acc1 = {};
    for (int kk = 0; kk < K; kk += 32) {
        if (kk + 32 < K) {               // WGP-scope next-tile prefetch
            __builtin_prefetch(Ap  + kk + 32, 0, 3);
            __builtin_prefetch(Bp0 + kk + 32, 0, 3);
            __builtin_prefetch(Bp1 + kk + 32, 0, 3);
        }
        v16bf a  = load_frag(Ap  + kk);
        v16bf b0 = load_frag(Bp0 + kk);
        v16bf b1 = load_frag(Bp1 + kk);
        acc0 = __builtin_amdgcn_wmma_f32_16x16x32_bf16(false, a, false, b0,
                                                       (short)0, acc0, false, false);
        acc1 = __builtin_amdgcn_wmma_f32_16x16x32_bf16(false, a, false, b1,
                                                       (short)0, acc1, false, false);
    }

    const int hi = (lane >> 4) & 1;
#pragma unroll
    for (int t = 0; t < 2; ++t) {
        const int n = n0 + t * 16 + (lane & 15);
        const float bn = bias ? bias[n] : 0.0f;
#pragma unroll
        for (int v = 0; v < 8; ++v) {
            int m = m0 + v + 8 * hi;
            float val = (t ? acc1[v] : acc0[v]) + bn;
            if (mode == 0) {
                outBF[(size_t)m * ldc + n] = f2bf(val);
            } else if (mode == 1) {
                // m = (b,t'), n = (l,d)  ->  Vt[((b*16+l)*64+d)*256 + t']
                int b_ = m >> 8, tt = m & 255, l = n >> 6, d = n & 63;
                outBF[((size_t)((b_ * 16 + l) * 64 + d)) * 256 + tt] = f2bf(val);
            } else {
                outF[(size_t)m * ldc + n] = val;
            }
        }
    }
}

// ---------------------------------------------------------------------------
// Kernel 3: attention scores  G[b,g,s,t] = scale*Q·K^T + spatial + edge
//   Per (b,g): [256x64]@[64x256].  Q,K stored flat [B*S, 512], head offset g*64.
//   Same 2-N-tile-per-wave scheme; block tile 32x128.
// ---------------------------------------------------------------------------
__global__ void scores_wmma(const __bf16* __restrict__ Q, const __bf16* __restrict__ Km,
                            const float* __restrict__ sp, const float* __restrict__ ed,
                            float* __restrict__ Gsc) {
    const int lane = threadIdx.x & 31;
    const int wid  = threadIdx.x >> 5;
    const int b = blockIdx.x >> 3, g = blockIdx.x & 7;
    const int m0 = blockIdx.y * 32  + (wid >> 2) * 16;
    const int n0 = blockIdx.z * 128 + (wid & 3) * 32;
    const int kb = (lane & 16) ? 8 : 0;

    const __bf16* Ap  = Q  + (size_t)(b * 256 + m0 + (lane & 15)) * 512 + g * 64 + kb;
    const __bf16* Bp0 = Km + (size_t)(b * 256 + n0 + (lane & 15)) * 512 + g * 64 + kb;
    const __bf16* Bp1 = Bp0 + (size_t)16 * 512;

    v8f acc0 = {}, acc1 = {};
#pragma unroll
    for (int kk = 0; kk < 64; kk += 32) {
        v16bf a  = load_frag(Ap  + kk);
        v16bf b0 = load_frag(Bp0 + kk);
        v16bf b1 = load_frag(Bp1 + kk);
        acc0 = __builtin_amdgcn_wmma_f32_16x16x32_bf16(false, a, false, b0,
                                                       (short)0, acc0, false, false);
        acc1 = __builtin_amdgcn_wmma_f32_16x16x32_bf16(false, a, false, b1,
                                                       (short)0, acc1, false, false);
    }

    const float scale = 0.04419417382415922f;   // 512^-0.5 (hidden_dim, not head)
    const int hi = (lane >> 4) & 1;
#pragma unroll
    for (int t = 0; t < 2; ++t) {
        const int n = n0 + t * 16 + (lane & 15);
#pragma unroll
        for (int v = 0; v < 8; ++v) {
            int m = m0 + v + 8 * hi;
            size_t bi = ((size_t)(b * 256 + m)) * 256 + n;
            Gsc[((size_t)((b * 8 + g) * 256 + m)) * 256 + n] =
                (t ? acc1[v] : acc0[v]) * scale + sp[bi] + ed[bi];
        }
    }
}

// ---------------------------------------------------------------------------
// Kernel 4: fused mix(G over 8 heads, diag sigma*eps, relu, pad) + softmax +
//           A@V.  One block per (b, l, 32-row tile).  G (16MB) is L2-resident,
//           so the 16x re-read over l is cheap.  A tile lives in LDS as bf16;
//           8 waves then each compute one 16x16 WMMA tile of [32x256]@[256x64].
// ---------------------------------------------------------------------------
__global__ void fused_attn(const float* __restrict__ Gsc, const __bf16* __restrict__ Vt,
                           const float* __restrict__ sigma, const float* __restrict__ p,
                           const float* __restrict__ epsd, __bf16* __restrict__ O) {
    __shared__ __align__(16) __bf16 Atile[32 * 272];   // 272 = 256 + pad (16B-mult stride)

    const int lane = threadIdx.x & 31;
    const int wid  = threadIdx.x >> 5;
    const int s0 = blockIdx.x * 32;
    const int l  = blockIdx.y;
    const int b  = blockIdx.z;

    // ---- Phase 1: one wave per row; lane covers columns lane+32j ----
    for (int it = 0; it < 4; ++it) {
        const int r = wid + it * 8;          // 0..31 local row
        const int s = s0 + r;                // row within batch [0,256)
        const float eps_s = epsd[l * 256 + s];

        float acc[8];
#pragma unroll
        for (int j = 0; j < 8; ++j) acc[j] = 0.0f;
        bool dead = false;

        for (int g = 0; g < 8; ++g) {
            const float* grow = Gsc + ((size_t)((b * 8 + g) * 256 + s)) * 256;
            const float pg = p[g * 16 + l];
            const float sg = sigma[g];
            const float se = sg * sg * eps_s;
            int nz = 0;
#pragma unroll
            for (int j = 0; j < 8; ++j) {
                int c = lane + 32 * j;
                float x = grow[c];
                nz |= (x != 0.0f);
                float t = (x + ((c == s) ? se : 0.0f)) * pg;
                acc[j] += fmaxf(t, 0.0f);
            }
            // pad[b,g,s] = any_t(G != 0); an all-zero row => -inf for this g
            // => whole summed row is -inf => softmax -> nan -> nan_to_num -> 0
            dead |= (__ballot(nz) == 0ull);
        }

        if (dead) {
#pragma unroll
            for (int j = 0; j < 8; ++j)
                Atile[r * 272 + lane + 32 * j] = f2bf(0.0f);
        } else {
            float m = acc[0];
#pragma unroll
            for (int j = 1; j < 8; ++j) m = fmaxf(m, acc[j]);
            for (int off = 16; off > 0; off >>= 1)
                m = fmaxf(m, __shfl_xor(m, off, 32));
            float e[8], ssum = 0.0f;
#pragma unroll
            for (int j = 0; j < 8; ++j) { e[j] = __expf(acc[j] - m); ssum += e[j]; }
            for (int off = 16; off > 0; off >>= 1)
                ssum += __shfl_xor(ssum, off, 32);
            float inv = 1.0f / ssum;
#pragma unroll
            for (int j = 0; j < 8; ++j)
                Atile[r * 272 + lane + 32 * j] = f2bf(e[j] * inv);
        }
    }
    __syncthreads();

    // ---- Phase 2: A[32x256] @ V[256x64] via 8 WMMA tiles (one per wave) ----
    const int m0 = (wid >> 2) * 16;          // 0 / 16
    const int n0 = (wid & 3) * 16;           // 0..48
    const int kb = (lane & 16) ? 8 : 0;
    const __bf16* Ap = Atile + (m0 + (lane & 15)) * 272 + kb;                 // LDS
    const __bf16* Vp = Vt + ((size_t)((b * 16 + l) * 64 + n0 + (lane & 15))) * 256 + kb;

    v8f acc2 = {};
#pragma unroll
    for (int kk = 0; kk < 256; kk += 32) {
        v16bf a  = load_frag(Ap + kk);
        v16bf bb = load_frag(Vp + kk);
        acc2 = __builtin_amdgcn_wmma_f32_16x16x32_bf16(false, a, false, bb,
                                                       (short)0, acc2, false, false);
    }

    const int hi = (lane >> 4) & 1;
    const int n  = n0 + (lane & 15);
#pragma unroll
    for (int v = 0; v < 8; ++v) {
        int srow = s0 + m0 + v + 8 * hi;
        O[((size_t)(b * 256 + srow)) * 1024 + l * 64 + n] = f2bf(acc2[v]);
    }
}

// ---------------------------------------------------------------------------
// Host-side launch
// ---------------------------------------------------------------------------
extern "C" void kernel_launch(void* const* d_in, const int* in_sizes, int n_in,
                              void* d_out, int out_size, void* d_ws, size_t ws_size,
                              hipStream_t stream) {
    (void)in_sizes; (void)n_in; (void)out_size; (void)ws_size;
    const float* x    = (const float*)d_in[0];
    const float* sp   = (const float*)d_in[1];
    const float* ed   = (const float*)d_in[2];
    const float* Wq   = (const float*)d_in[3];
    const float* Wk   = (const float*)d_in[4];
    const float* Wv   = (const float*)d_in[5];
    const float* bv   = (const float*)d_in[6];
    const float* sig  = (const float*)d_in[7];
    const float* pmat = (const float*)d_in[8];
    const float* epsd = (const float*)d_in[9];
    const float* Wout = (const float*)d_in[10];
    const float* bout = (const float*)d_in[11];

    char* ws = (char*)d_ws;
    size_t off = 0;
    auto alloc = [&](size_t bytes) -> void* {
        void* r = ws + off;
        off += (bytes + 255) & ~(size_t)255;
        return r;
    };
    __bf16* xbf   = (__bf16*)alloc(2048ull * 512 * 2);
    __bf16* WqT   = (__bf16*)alloc(512ull * 512 * 2);
    __bf16* WkT   = (__bf16*)alloc(512ull * 512 * 2);
    __bf16* WvT   = (__bf16*)alloc(1024ull * 512 * 2);   // [N=1024][K=512]
    __bf16* WoutT = (__bf16*)alloc(512ull * 1024 * 2);   // [N=512][K=1024]
    __bf16* Qbf   = (__bf16*)alloc(2048ull * 512 * 2);
    __bf16* Kbf   = (__bf16*)alloc(2048ull * 512 * 2);
    __bf16* Vt    = (__bf16*)alloc(8ull * 16 * 64 * 256 * 2);  // [B][L][d][t]
    float*  Gsc   = (float*)alloc(8ull * 8 * 256 * 256 * 4);   // 16MB (L2-resident)
    __bf16* Obf   = (__bf16*)alloc(2048ull * 1024 * 2);

    dim3 blk(256);

    // 1) bf16 conversions (+ weight transposes)
    cvt_bf16   <<<dim3((1048576 + 255) / 256), blk, 0, stream>>>(x, xbf, 1048576);
    cvt_t_bf16 <<<dim3((262144 + 255) / 256), blk, 0, stream>>>(Wq, WqT, 512, 512);
    cvt_t_bf16 <<<dim3((262144 + 255) / 256), blk, 0, stream>>>(Wk, WkT, 512, 512);
    cvt_t_bf16 <<<dim3((524288 + 255) / 256), blk, 0, stream>>>(Wv, WvT, 512, 1024);
    cvt_t_bf16 <<<dim3((524288 + 255) / 256), blk, 0, stream>>>(Wout, WoutT, 1024, 512);

    // 2) projections: Q = x@Wq, K = x@Wk (bf16 out), V = x@Wv+bv (transposed)
    gemm_bf16_wmma<<<dim3(64, 4), blk, 0, stream>>>(xbf, 512, WqT, 512, 512,
                                                    nullptr, nullptr, Qbf, 0, 512);
    gemm_bf16_wmma<<<dim3(64, 4), blk, 0, stream>>>(xbf, 512, WkT, 512, 512,
                                                    nullptr, nullptr, Kbf, 0, 512);
    gemm_bf16_wmma<<<dim3(64, 8), blk, 0, stream>>>(xbf, 512, WvT, 512, 512,
                                                    bv, nullptr, Vt, 1, 0);

    // 3) scores G = scale*QK^T + spatial + edge   (f32, 16MB)
    scores_wmma<<<dim3(64, 8, 2), blk, 0, stream>>>(Qbf, Kbf, sp, ed, Gsc);

    // 4) fused head-mix + softmax + A@V   -> Obf [B*S, L*64] bf16
    fused_attn<<<dim3(8, 16, 8), blk, 0, stream>>>(Gsc, Vt, sig, pmat, epsd, Obf);

    // 5) final: out = Obf @ Wout + bout   (f32 to d_out)
    gemm_bf16_wmma<<<dim3(64, 4), blk, 0, stream>>>(Obf, 1024, WoutT, 1024, 1024,
                                                    bout, (float*)d_out, nullptr, 2, 512);
}
